// AttentionPairBias_54365696033085
// MI455X (gfx1250) — compile-verified, matrix operand order checked by reading, not run on hardware
//
#include <hip/hip_runtime.h>
#include <hip/hip_bf16.h>

// ---------------------------------------------------------------------------
// AttentionPairBias for MI455X (gfx1250, wave32, WMMA 16x16x32 bf16)
// B=1, N=1024, C_A=768, C_Z=128, H=16, CH=48 (padded to 64 for WMMA K-loop)
// ---------------------------------------------------------------------------

typedef __attribute__((ext_vector_type(16))) __bf16 v16bf;
typedef __attribute__((ext_vector_type(8)))  __bf16 v8bf;
typedef __attribute__((ext_vector_type(8)))  float  v8f;

using u16 = unsigned short;

static constexpr int   NN   = 1024;
static constexpr int   CA   = 768;
static constexpr int   CZ   = 128;
static constexpr int   NH   = 16;
static constexpr int   CH   = 48;
static constexpr int   CHP  = 64;
static constexpr float INV_SQRT_CH = 0.14433756729740643f;

// float -> bf16 bits, round-to-nearest-even
__device__ __forceinline__ u16 f2bf(float f) {
  unsigned int u = __float_as_uint(f);
  u += 0x7fffu + ((u >> 16) & 1u);
  return (u16)(u >> 16);
}
__device__ __forceinline__ __bf16 tobf(float f) {
  u16 s = f2bf(f);
  __bf16 b;
  __builtin_memcpy(&b, &s, 2);
  return b;
}

// Load a 16x32 bf16 tile in the documented wave32 16-bit A/B WMMA layout.
// lane m = lane&15 selects row (M for A, N for B); kh = lane>>4 selects the
// two 8-element contiguous K segments: [8*kh, 8*kh+8) and [16+8*kh, 24+8*kh).
__device__ __forceinline__ v16bf ld_tile16x32(const u16* base, int ld, int row0, int col0) {
  int lane = threadIdx.x & 31;
  int m = lane & 15, kh = (lane >> 4) & 1;
  const __bf16* p = reinterpret_cast<const __bf16*>(base)
                  + (size_t)(row0 + m) * ld + col0 + kh * 8;
  v8bf lo = *reinterpret_cast<const v8bf*>(p);
  v8bf hi = *reinterpret_cast<const v8bf*>(p + 16);
  v16bf r;
#pragma unroll
  for (int i = 0; i < 8; ++i) { r[i] = lo[i]; r[i + 8] = hi[i]; }
  return r;
}

__device__ __forceinline__ v8f wmma_bf16(v16bf a, v16bf b, v8f c) {
  return __builtin_amdgcn_wmma_f32_16x16x32_bf16(false, a, false, b, (short)0, c, false, false);
}

// ---------------------------------------------------------------------------
// 0) zero the padded q/k buffers (cols 48..63 must be zero)
// ---------------------------------------------------------------------------
__global__ void k_zero(unsigned int* p, size_t nwords) {
  size_t i = (size_t)blockIdx.x * blockDim.x + threadIdx.x;
  size_t step = (size_t)gridDim.x * blockDim.x;
  for (size_t k = i; k < nwords; k += step) p[k] = 0u;
}

// ---------------------------------------------------------------------------
// 1) convert + transpose the five 768x768 fp32 weights -> bf16 [out][in]
// ---------------------------------------------------------------------------
__global__ __launch_bounds__(256) void k_cvt_w(const float* wq, const float* wk,
                                               const float* wv, const float* wg,
                                               const float* wo, u16* wt_all) {
  __shared__ float tile[32][33];
  int which = blockIdx.z;
  const float* src = (which == 0) ? wq : (which == 1) ? wk : (which == 2) ? wv
                     : (which == 3) ? wg : wo;
  u16* dst = wt_all + (size_t)which * CA * CA;
  int tx = threadIdx.x, ty = threadIdx.y;           // blockDim = (32, 8)
#pragma unroll
  for (int j = 0; j < 4; ++j) {
    int i = blockIdx.y * 32 + ty + j * 8;           // input row
    int o = blockIdx.x * 32 + tx;                   // input col
    tile[ty + j * 8][tx] = src[(size_t)i * CA + o];
  }
  __syncthreads();
#pragma unroll
  for (int j = 0; j < 4; ++j) {
    int o = blockIdx.x * 32 + ty + j * 8;
    int i = blockIdx.y * 32 + tx;
    dst[(size_t)o * CA + i] = f2bf(tile[tx][ty + j * 8]);
  }
}

// ---------------------------------------------------------------------------
// 2) fold ln_z into Wb:  Wb2t[h][c] = ln_z_w[c]*Wb[c][h] (bf16),
//    colsum[h] = sum_c Wb2, biash[h] = sum_c ln_z_b[c]*Wb[c][h]
// ---------------------------------------------------------------------------
__global__ __launch_bounds__(256) void k_prep_wb(const float* lnzw, const float* lnzb,
                                                 const float* Wb, u16* wb2t,
                                                 float* colsum, float* biash) {
  int t = threadIdx.x;
  for (int idx = t; idx < NH * CZ; idx += 256) {
    int h = idx >> 7, c = idx & 127;
    wb2t[h * CZ + c] = f2bf(lnzw[c] * Wb[c * NH + h]);
  }
  if (t < NH) {
    float cs = 0.f, bh = 0.f;
    for (int c = 0; c < CZ; ++c) {
      float wbv = Wb[c * NH + t];
      cs += lnzw[c] * wbv;
      bh += lnzb[c] * wbv;
    }
    colsum[t] = cs;
    biash[t] = bh;
  }
}

// ---------------------------------------------------------------------------
// 3) LayerNorm(a) -> bf16 [1024][768]
// ---------------------------------------------------------------------------
__global__ __launch_bounds__(256) void k_ln_a(const float* a, const float* w,
                                              const float* b, u16* an) {
  __shared__ float red[256];
  int row = blockIdx.x, t = threadIdx.x;
  const float* ar = a + (size_t)row * CA;
  float x0 = ar[t], x1 = ar[t + 256], x2 = ar[t + 512];
  red[t] = x0 + x1 + x2;
  __syncthreads();
  for (int o = 128; o > 0; o >>= 1) { if (t < o) red[t] += red[t + o]; __syncthreads(); }
  float mean = red[0] * (1.f / CA);
  __syncthreads();
  float d0 = x0 - mean, d1 = x1 - mean, d2 = x2 - mean;
  red[t] = d0 * d0 + d1 * d1 + d2 * d2;
  __syncthreads();
  for (int o = 128; o > 0; o >>= 1) { if (t < o) red[t] += red[t + o]; __syncthreads(); }
  float rs = rsqrtf(red[0] * (1.f / CA) + 1e-5f);
  u16* anr = an + (size_t)row * CA;
  anr[t]       = f2bf(d0 * rs * w[t]       + b[t]);
  anr[t + 256] = f2bf(d1 * rs * w[t + 256] + b[t + 256]);
  anr[t + 512] = f2bf(d2 * rs * w[t + 512] + b[t + 512]);
}

// ---------------------------------------------------------------------------
// 4) projections: an @ {Wq,Wk,Wv,Wg}; wave per 32x32 output (2x2 tiles so
//    each A/B tile load feeds two WMMAs -> half the L2 traffic), K=768.
//    mode 0: q = (.+bq)*inv_sqrt -> qpad[h][n][64]   (bf16, pre-scaled)
//    mode 1: k                   -> kpad[h][n][64]
//    mode 2: v                   -> vT[h*48+c][n]
//    mode 3: g -> sigmoid fp32   -> gsig[n][768]
// ---------------------------------------------------------------------------
__global__ __launch_bounds__(128) void k_gemm_qkvg(const u16* an, const u16* wt_all,
                                                   const float* bq, u16* qpad, u16* kpad,
                                                   u16* vt, float* gsig) {
  int wid = threadIdx.x >> 5, lane = threadIdx.x & 31;
  int mode = blockIdx.y;
  int gw = blockIdx.x * 4 + wid;           // 0..767
  int tm2 = gw / 24, tn2 = gw % 24;        // 32x32 super-tile
  const u16* wt = wt_all + (size_t)mode * CA * CA;
  v8f acc[2][2] = {};
  for (int kk = 0; kk < CA; kk += 32) {
    v16bf a0 = ld_tile16x32(an, CA, tm2 * 32,      kk);
    v16bf a1 = ld_tile16x32(an, CA, tm2 * 32 + 16, kk);
    v16bf b0 = ld_tile16x32(wt, CA, tn2 * 32,      kk);
    v16bf b1 = ld_tile16x32(wt, CA, tn2 * 32 + 16, kk);
    acc[0][0] = wmma_bf16(a0, b0, acc[0][0]);
    acc[0][1] = wmma_bf16(a0, b1, acc[0][1]);
    acc[1][0] = wmma_bf16(a1, b0, acc[1][0]);
    acc[1][1] = wmma_bf16(a1, b1, acc[1][1]);
  }
  int hh = (lane >> 4) & 1;
#pragma unroll
  for (int ti = 0; ti < 2; ++ti) {
#pragma unroll
    for (int tj = 0; tj < 2; ++tj) {
      int col = tn2 * 32 + tj * 16 + (lane & 15);
      int h = col / CH, cc = col % CH;
#pragma unroll
      for (int r = 0; r < 8; ++r) {
        int row = tm2 * 32 + ti * 16 + r + 8 * hh;
        float v = acc[ti][tj][r];
        if (mode == 0) {
          v = (v + bq[col]) * INV_SQRT_CH;
          qpad[((size_t)h * NN + row) * CHP + cc] = f2bf(v);
        } else if (mode == 1) {
          kpad[((size_t)h * NN + row) * CHP + cc] = f2bf(v);
        } else if (mode == 2) {
          vt[((size_t)(h * CH + cc)) * NN + row] = f2bf(v);
        } else {
          gsig[(size_t)row * CA + col] = 1.0f / (1.0f + __expf(-v));
        }
      }
    }
  }
}

// ---------------------------------------------------------------------------
// 5) pair bias: one pass over z (537 MB -> the bandwidth-critical kernel).
//    Wave handles 16 pairs x 128 ch: loads raw z directly in WMMA A-layout,
//    computes per-pair mean/var from registers, 4 WMMAs vs folded Wb2,
//    fixes up with rs/mean/colsum/biash and scatters to bpair[h][q][k].
// ---------------------------------------------------------------------------
__global__ __launch_bounds__(128) void k_pairbias(const float* z, const u16* wb2t,
                                                  const float* colsum, const float* biash,
                                                  float* bpair) {
  __shared__ float2 mrs[4][16];
  int wid = threadIdx.x >> 5, lane = threadIdx.x & 31;
  int gw = blockIdx.x * 4 + wid;
  int qi = gw >> 6, kt = gw & 63;
  int k0 = kt * 16;
  int m = lane & 15, kh = (lane >> 4) & 1, hh = kh;
  const float* zrow = z + ((size_t)qi * NN + (k0 + m)) * CZ;
  float zv[64];
#pragma unroll
  for (int q = 0; q < 4; ++q) {
    const float* p = zrow + q * 32 + kh * 8;
    float4 a0 = *(const float4*)(p);
    float4 a1 = *(const float4*)(p + 4);
    float4 b0 = *(const float4*)(p + 16);
    float4 b1 = *(const float4*)(p + 20);
    zv[q*16+0]=a0.x; zv[q*16+1]=a0.y; zv[q*16+2]=a0.z; zv[q*16+3]=a0.w;
    zv[q*16+4]=a1.x; zv[q*16+5]=a1.y; zv[q*16+6]=a1.z; zv[q*16+7]=a1.w;
    zv[q*16+8]=b0.x; zv[q*16+9]=b0.y; zv[q*16+10]=b0.z; zv[q*16+11]=b0.w;
    zv[q*16+12]=b1.x; zv[q*16+13]=b1.y; zv[q*16+14]=b1.z; zv[q*16+15]=b1.w;
  }
  float s1 = 0.f, s2 = 0.f;
#pragma unroll
  for (int i = 0; i < 64; ++i) { s1 += zv[i]; s2 += zv[i] * zv[i]; }
  s1 += __shfl_xor(s1, 16);
  s2 += __shfl_xor(s2, 16);
  float mean = s1 * (1.f / CZ);
  float var  = s2 * (1.f / CZ) - mean * mean;
  float rs = rsqrtf(var + 1e-5f);
  if (lane < 16) mrs[wid][lane] = make_float2(mean, rs);
  v8f acc = {};
#pragma unroll
  for (int q = 0; q < 4; ++q) {
    v16bf av;
#pragma unroll
    for (int j = 0; j < 16; ++j) av[j] = tobf(zv[q * 16 + j]);
    v16bf bv = ld_tile16x32(wb2t, CZ, 0, q * 32);
    acc = wmma_bf16(av, bv, acc);
  }
  int hcol = lane & 15;
  float cs = colsum[hcol], bh = biash[hcol];
#pragma unroll
  for (int r = 0; r < 8; ++r) {
    int p = r + 8 * hh;
    float2 mr = mrs[wid][p];
    float val = mr.y * (acc[r] - mr.x * cs) + bh;
    bpair[((size_t)hcol * NN + qi) * NN + (k0 + p)] = val;
  }
}

// ---------------------------------------------------------------------------
// 6) flash attention: wave per (head, 32-query group) -- two 16-query tiles
//    share every K/V tile load (halves K/V L2 traffic vs 16-query waves).
//    logits: 4 WMMAs / (16q x 32 keys); row stats via shfl over the 16-lane
//    half-group (C-layout: N=key across lanes 0..15); P -> A-layout through
//    LDS; PV: 3 WMMAs / (16q x 32 keys) into 16x48 f32 accumulators.
// ---------------------------------------------------------------------------
__global__ __launch_bounds__(128) void k_attn(const u16* qpad, const u16* kpad,
                                              const u16* vt, const float* bpair,
                                              const float* mask, const float* gsig,
                                              u16* go) {
  __shared__ float pbuf[4][2 * 16 * 32];
  int wid = threadIdx.x >> 5, lane = threadIdx.x & 31;
  int gw = blockIdx.x * 4 + wid;           // 0..511
  int h = gw >> 5, qt = gw & 31;           // 32 query-groups of 32
  int q0 = qt * 32;
  int hh = (lane >> 4) & 1, ln16 = lane & 15;
  const u16* qb = qpad + (size_t)h * NN * CHP;
  const u16* kb = kpad + (size_t)h * NN * CHP;
  v16bf qa[2][2];
#pragma unroll
  for (int s = 0; s < 2; ++s) {
    qa[s][0] = ld_tile16x32(qb, CHP, q0 + s * 16, 0);
    qa[s][1] = ld_tile16x32(qb, CHP, q0 + s * 16, 32);
  }
  v8f o[2][3] = {};
  float mrun[2][8], lrun[2][8];
#pragma unroll
  for (int s = 0; s < 2; ++s)
#pragma unroll
    for (int r = 0; r < 8; ++r) { mrun[s][r] = -1e30f; lrun[s][r] = 0.f; }
  const float* bp = bpair + ((size_t)h * NN + q0) * NN;
  float* pl = pbuf[wid];
  for (int k0 = 0; k0 < NN; k0 += 32) {
    v16bf b00 = ld_tile16x32(kb, CHP, k0, 0);
    v16bf b01 = ld_tile16x32(kb, CHP, k0, 32);
    v16bf b10 = ld_tile16x32(kb, CHP, k0 + 16, 0);
    v16bf b11 = ld_tile16x32(kb, CHP, k0 + 16, 32);
    float mb0 = (mask[k0 + ln16] - 1.0f) * 1e9f;
    float mb1 = (mask[k0 + 16 + ln16] - 1.0f) * 1e9f;
#pragma unroll
    for (int s = 0; s < 2; ++s) {
      v8f s0 = {}, s1 = {};
      s0 = wmma_bf16(qa[s][0], b00, s0);
      s0 = wmma_bf16(qa[s][1], b01, s0);
      s1 = wmma_bf16(qa[s][0], b10, s1);
      s1 = wmma_bf16(qa[s][1], b11, s1);
      float* pls = pl + s * 512;
      const float* bps = bp + (size_t)s * 16 * NN;
#pragma unroll
      for (int r = 0; r < 8; ++r) {
        int row = r + 8 * hh;
        float v0 = s0[r] + bps[(size_t)row * NN + k0 + ln16] + mb0;
        float v1 = s1[r] + bps[(size_t)row * NN + k0 + 16 + ln16] + mb1;
        float mx = fmaxf(v0, v1);
        mx = fmaxf(mx, __shfl_xor(mx, 1));
        mx = fmaxf(mx, __shfl_xor(mx, 2));
        mx = fmaxf(mx, __shfl_xor(mx, 4));
        mx = fmaxf(mx, __shfl_xor(mx, 8));
        float mnew = fmaxf(mrun[s][r], mx);
        float sc = __expf(mrun[s][r] - mnew);
        float p0 = __expf(v0 - mnew);
        float p1 = __expf(v1 - mnew);
        float ps = p0 + p1;
        ps += __shfl_xor(ps, 1);
        ps += __shfl_xor(ps, 2);
        ps += __shfl_xor(ps, 4);
        ps += __shfl_xor(ps, 8);
        lrun[s][r] = lrun[s][r] * sc + ps;
        mrun[s][r] = mnew;
        o[s][0][r] *= sc; o[s][1][r] *= sc; o[s][2][r] *= sc;
        pls[row * 32 + ln16] = p0;
        pls[row * 32 + 16 + ln16] = p1;
      }
    }
    v16bf vb0 = ld_tile16x32(vt, NN, h * CH + 0,  k0);
    v16bf vb1 = ld_tile16x32(vt, NN, h * CH + 16, k0);
    v16bf vb2 = ld_tile16x32(vt, NN, h * CH + 32, k0);
#pragma unroll
    for (int s = 0; s < 2; ++s) {
      const float* pls = pl + s * 512;
      v16bf pa;
#pragma unroll
      for (int j = 0; j < 8; ++j) {
        pa[j]     = tobf(pls[ln16 * 32 + hh * 8 + j]);
        pa[j + 8] = tobf(pls[ln16 * 32 + 16 + hh * 8 + j]);
      }
      o[s][0] = wmma_bf16(pa, vb0, o[s][0]);
      o[s][1] = wmma_bf16(pa, vb1, o[s][1]);
      o[s][2] = wmma_bf16(pa, vb2, o[s][2]);
    }
  }
#pragma unroll
  for (int s = 0; s < 2; ++s) {
#pragma unroll
    for (int r = 0; r < 8; ++r) {
      int row = q0 + s * 16 + r + 8 * hh;
      float inv = 1.0f / lrun[s][r];
      size_t base = (size_t)row * CA + h * CH + ln16;
      go[base]      = f2bf(o[s][0][r] * inv * gsig[base]);
      go[base + 16] = f2bf(o[s][1][r] * inv * gsig[base + 16]);
      go[base + 32] = f2bf(o[s][2][r] * inv * gsig[base + 32]);
    }
  }
}

// ---------------------------------------------------------------------------
// 7) output projection: go(bf16) @ Wo -> d_out fp32 (same 2x2 blocking)
// ---------------------------------------------------------------------------
__global__ __launch_bounds__(128) void k_gemm_out(const u16* go, const u16* wto, float* out) {
  int wid = threadIdx.x >> 5, lane = threadIdx.x & 31;
  int gw = blockIdx.x * 4 + wid;           // 0..767
  int tm2 = gw / 24, tn2 = gw % 24;
  v8f acc[2][2] = {};
  for (int kk = 0; kk < CA; kk += 32) {
    v16bf a0 = ld_tile16x32(go,  CA, tm2 * 32,      kk);
    v16bf a1 = ld_tile16x32(go,  CA, tm2 * 32 + 16, kk);
    v16bf b0 = ld_tile16x32(wto, CA, tn2 * 32,      kk);
    v16bf b1 = ld_tile16x32(wto, CA, tn2 * 32 + 16, kk);
    acc[0][0] = wmma_bf16(a0, b0, acc[0][0]);
    acc[0][1] = wmma_bf16(a0, b1, acc[0][1]);
    acc[1][0] = wmma_bf16(a1, b0, acc[1][0]);
    acc[1][1] = wmma_bf16(a1, b1, acc[1][1]);
  }
  int hh = (lane >> 4) & 1;
#pragma unroll
  for (int ti = 0; ti < 2; ++ti) {
#pragma unroll
    for (int tj = 0; tj < 2; ++tj) {
      int col = tn2 * 32 + tj * 16 + (lane & 15);
#pragma unroll
      for (int r = 0; r < 8; ++r) {
        int row = tm2 * 32 + ti * 16 + r + 8 * hh;
        out[(size_t)row * CA + col] = acc[ti][tj][r];
      }
    }
  }
}

// ---------------------------------------------------------------------------
// launch
// ---------------------------------------------------------------------------
extern "C" void kernel_launch(void* const* d_in, const int* in_sizes, int n_in,
                              void* d_out, int out_size, void* d_ws, size_t ws_size,
                              hipStream_t stream) {
  const float* a     = (const float*)d_in[0];
  const float* z     = (const float*)d_in[1];
  const float* mask  = (const float*)d_in[2];
  const float* ln_aw = (const float*)d_in[3];
  const float* ln_ab = (const float*)d_in[4];
  const float* ln_zw = (const float*)d_in[5];
  const float* ln_zb = (const float*)d_in[6];
  const float* Wq    = (const float*)d_in[7];
  const float* bq    = (const float*)d_in[8];
  const float* Wk    = (const float*)d_in[9];
  const float* Wv    = (const float*)d_in[10];
  const float* Wb    = (const float*)d_in[11];
  const float* Wg    = (const float*)d_in[12];
  const float* Wo    = (const float*)d_in[13];
  float* out = (float*)d_out;

  char* ws = (char*)d_ws;
  size_t off = 0;
  auto alloc = [&](size_t bytes) { size_t o = off; off += (bytes + 255) & ~(size_t)255; return o; };
  size_t off_wt    = alloc((size_t)5 * CA * CA * 2);   // 5 bf16 weights [out][in]
  size_t off_an    = alloc((size_t)NN * CA * 2);       // LN(a) bf16
  size_t off_qpad  = alloc((size_t)NH * NN * CHP * 2); // q bf16 padded
  size_t off_kpad  = alloc((size_t)NH * NN * CHP * 2); // k bf16 padded
  size_t off_vt    = alloc((size_t)NH * CH * NN * 2);  // v^T bf16
  size_t off_gsig  = alloc((size_t)NN * CA * 4);       // sigmoid gate fp32
  size_t off_go    = alloc((size_t)NN * CA * 2);       // gated o bf16
  size_t off_wb2t  = alloc((size_t)NH * CZ * 2);       // folded Wb bf16 [h][c]
  size_t off_cs    = alloc((size_t)NH * 4);
  size_t off_bh    = alloc((size_t)NH * 4);
  size_t off_bpair = alloc((size_t)NH * NN * NN * 4);  // pair bias fp32 [h][q][k]

  u16*   wt_all = (u16*)(ws + off_wt);
  u16*   an     = (u16*)(ws + off_an);
  u16*   qpad   = (u16*)(ws + off_qpad);
  u16*   kpad   = (u16*)(ws + off_kpad);
  u16*   vt     = (u16*)(ws + off_vt);
  float* gsig   = (float*)(ws + off_gsig);
  u16*   go     = (u16*)(ws + off_go);
  u16*   wb2t   = (u16*)(ws + off_wb2t);
  float* cs     = (float*)(ws + off_cs);
  float* bh     = (float*)(ws + off_bh);
  float* bpair  = (float*)(ws + off_bpair);

  // 0) zero the padded q/k buffers (contiguous region)
  size_t padwords = ((size_t)NH * NN * CHP * 2 * 2) / 4;
  k_zero<<<1024, 256, 0, stream>>>((unsigned int*)qpad, padwords);

  // 1) weights -> bf16 [out][in]
  k_cvt_w<<<dim3(24, 24, 5), dim3(32, 8), 0, stream>>>(Wq, Wk, Wv, Wg, Wo, wt_all);

  // 2) fold ln_z into Wb
  k_prep_wb<<<1, 256, 0, stream>>>(ln_zw, ln_zb, Wb, wb2t, cs, bh);

  // 3) LN(a)
  k_ln_a<<<NN, 256, 0, stream>>>(a, ln_aw, ln_ab, an);

  // 4) q,k,v,g projections (32x32 per wave, 4 waves/block, grid.y = mode)
  k_gemm_qkvg<<<dim3(192, 4), 128, 0, stream>>>(an, wt_all, bq, qpad, kpad, vt, gsig);

  // 5) pair bias from z (single pass, WMMA)
  k_pairbias<<<16384, 128, 0, stream>>>(z, wb2t, cs, bh, bpair);

  // 6) flash attention + gating (32 queries per wave)
  k_attn<<<128, 128, 0, stream>>>(qpad, kpad, vt, bpair, mask, gsig, go);

  // 7) output projection
  k_gemm_out<<<192, 128, 0, stream>>>(go, wt_all + (size_t)4 * CA * CA, out);

  (void)in_sizes; (void)n_in; (void)out_size; (void)ws_size;
}
